// AudioTransformer_43748536877702
// MI455X (gfx1250) — matrix operand order repaired and checked by reading, not compiled, not run on hardware
//
#include <hip/hip_runtime.h>
#include <math.h>

// ---------------------------------------------------------------------------
// AudioTransformer for MI455X (gfx1250): L=512 tokens, D=512, 8 layers.
//  * GEMMs: v_wmma_f32_16x16x32_f16, f32 accumulate.
//  * Weights pre-packed once per launch to f16 fragment-major (48MB, L2-resident).
//  * A tiles staged to LDS by the Tensor Data Mover (tensor_load_to_lds),
//    double-buffered, completion via s_wait_tensorcnt + barrier.
// ---------------------------------------------------------------------------

#define DIM     512
#define LAYERS  8
#define PATCH   32
#define HEADS   8
#define DH      64
#define KWIN    127
#define SEQ     512      // 16384 / 32
#define FFDIM   2048     // 4 * DIM

typedef __attribute__((ext_vector_type(16))) _Float16 v16h;
typedef __attribute__((ext_vector_type(8)))  _Float16 v8h;
typedef __attribute__((ext_vector_type(8)))  float    v8f;
typedef __attribute__((ext_vector_type(4)))  unsigned int v4u;
typedef __attribute__((ext_vector_type(8)))  int          v8i;
typedef __attribute__((ext_vector_type(4)))  int          v4i;

#if defined(__has_builtin)
#  if __has_builtin(__builtin_amdgcn_tensor_load_to_lds) && \
      __has_builtin(__builtin_amdgcn_s_wait_tensorcnt)
#    define USE_TDM 1
#  endif
#endif
#ifndef USE_TDM
#  define USE_TDM 0
#endif

// ---------------------------------------------------------------------------
// TDM descriptor (CDNA5 ISA 08_async_tensor §8): load a 2-D tile of
// 16 dwords x 64 rows (64x32 f16) from a row-major tensor into LDS.
// ---------------------------------------------------------------------------
#if USE_TDM
__device__ inline void tdm_load_tile64x32(const _Float16* gsrc, unsigned ldsOff,
                                          int rowLenDw, int nRows) {
    const unsigned long long ga = (unsigned long long)(size_t)gsrc;
    v4u g0;
    g0[0] = 1u;                                     // count=1 (valid), user mode
    g0[1] = ldsOff;                                 // lds_addr [63:32]
    g0[2] = (unsigned)(ga & 0xFFFFFFFFu);           // global_addr [95:64]
    g0[3] = (unsigned)((ga >> 32) & 0x01FFFFFFu)    // global_addr [120:96]
          | 0x80000000u;                            // type=2 ("image") [127:126]
    v8i g1 = {};
    g1[0] = (2 << 16);                              // data_size=2 -> 4 bytes
    g1[1] = (rowLenDw & 0xFFFF) << 16;              // tensor_dim0 [79:48] lo
    g1[2] = ((rowLenDw >> 16) & 0xFFFF)             // tensor_dim0 hi
          | ((nRows & 0xFFFF) << 16);               // tensor_dim1 [111:80] lo
    g1[3] = ((nRows >> 16) & 0xFFFF)                // tensor_dim1 hi
          | (16 << 16);                             // tile_dim0 = 16 dwords
    g1[4] = 64;                                     // tile_dim1 = 64 rows, dim2=0
    g1[5] = rowLenDw;                               // tensor_dim0_stride lo32
    g1[6] = 0;                                      // stride0 hi / stride1 lo
    g1[7] = 0;
    v4i g2 = {};
    v4i g3 = {};
#if __clang_major__ >= 23
    v8i g4 = {};
    __builtin_amdgcn_tensor_load_to_lds(g0, g1, g2, g3, g4, 0);
#else
    __builtin_amdgcn_tensor_load_to_lds(g0, g1, g2, g3, 0);
#endif
}
#endif

// ---------------------------------------------------------------------------
// WMMA fragment helpers (layouts per CDNA5 ISA 7.12.2, wave32)
// ---------------------------------------------------------------------------

// A fragment from LDS tile (row-major 64x32 f16): row m = m0 + lane%16.
// lanes 0-15 : VGPR0-3 = K 0..7,  VGPR4-7 = K 16..23
// lanes 16-31: VGPR0-3 = K 8..15, VGPR4-7 = K 24..31
__device__ inline v16h lds_a_frag(const _Float16* Ab, int m0, int lane) {
    const int m  = m0 + (lane & 15);
    const int kb = (lane >> 4) << 3;            // +0 or +8
    const _Float16* p = Ab + m * 32 + kb;
    v8h lo = *(const v8h*)p;                    // K = kb .. kb+7
    v8h hi = *(const v8h*)(p + 16);             // K = kb+16 .. kb+23
    v16h r;
#pragma unroll
    for (int i = 0; i < 8; ++i) { r[i] = lo[i]; r[8 + i] = hi[i]; }
    return r;
}

__device__ inline v8f wmma_f16(v16h a, v16h b, v8f c) {
    return __builtin_amdgcn_wmma_f32_16x16x32_f16(false, a, false, b,
                                                  (short)0, c, false, false);
}

// ---------------------------------------------------------------------------
// Weight pre-pack: W[N][K] f32 row-major  ->  PW fragment-major f16:
// PW[nt][kc][lane][i] = W[nt*16 + lane%16][kc*32 + (lane<16?0:8) + (i<8?i:i+8)]
// so a B fragment is one contiguous 32B load per lane (1KB/wave, coalesced).
// ---------------------------------------------------------------------------
__global__ __launch_bounds__(256) void pack_w_k(const float* __restrict__ W,
                                                _Float16* __restrict__ PW,
                                                int N, int Kd) {
    const int idx  = blockIdx.x * 256 + threadIdx.x;   // [0, N*Kd/16)
    const int lane = idx & 31;
    const int nkc  = Kd >> 5;
    const int kc   = (idx >> 5) % nkc;
    const int nt   = (idx >> 5) / nkc;
    const int n    = nt * 16 + (lane & 15);
    const int kb   = kc * 32 + ((lane >> 4) << 3);
    const float* src = W + (size_t)n * Kd + kb;
    _Float16* dst = PW + (size_t)idx * 16;
#pragma unroll
    for (int i = 0; i < 16; ++i) dst[i] = (_Float16)src[i < 8 ? i : i + 8];
}

// ---------------------------------------------------------------------------
// WMMA GEMM:  Y[M,N] = A[M,K](f16) @ W(f16 packed).T + bias
// Block = 256 thr = 8 waves; wave tile 32x32 (2x2 WMMA); block tile 64x128.
// A tile (64x32 f16 = 4KB) staged to LDS by TDM, double-buffered.
// EPI: 0 = store f32, 1 = residual add into f32, 2 = exact GELU -> f16 store
// ---------------------------------------------------------------------------
template <int EPI>
__global__ __launch_bounds__(256) void gemm_wmma(
        const _Float16* __restrict__ A, const _Float16* __restrict__ PB,
        const float* __restrict__ bias, float* __restrict__ Yf,
        _Float16* __restrict__ Yh, int M, int N, int Kd) {
    __shared__ __attribute__((aligned(128))) _Float16 As[2][64 * 32];
    const int tid  = threadIdx.x;
    const int lane = tid & 31;
    const int wave = tid >> 5;
    const int mblk = blockIdx.y * 64;
    const int m0l  = (wave >> 2) * 32;                 // local row offset
    const int n0   = blockIdx.x * 128 + (wave & 3) * 32;
    const int nk   = Kd >> 5;

    v8f acc[2][2];
#pragma unroll
    for (int ni = 0; ni < 2; ++ni) {
        const float bv = bias[n0 + ni * 16 + (lane & 15)];
#pragma unroll
        for (int mi = 0; mi < 2; ++mi)
#pragma unroll
            for (int r = 0; r < 8; ++r) acc[mi][ni][r] = bv;
    }

    // packed-B fragment base for this wave's two n-tiles at k-chunk 0
    const _Float16* bp0 = PB + (((size_t)(n0 >> 4)) * nk << 9) + (lane << 4);
    const size_t bTileStride = (size_t)nk << 9;        // next n-tile (+16 cols)

#if USE_TDM
    const _Float16* aTile = A + (size_t)mblk * Kd;     // tile row 0, chunk 0
    if (wave == 0) {
        tdm_load_tile64x32(aTile, (unsigned)(size_t)&As[0][0], Kd >> 1, M);
    }
    for (int c = 0; c < nk; ++c) {
        if (wave == 0 && c + 1 < nk) {
            tdm_load_tile64x32(aTile + (c + 1) * 32,
                               (unsigned)(size_t)&As[(c + 1) & 1][0], Kd >> 1, M);
        }
        // SIMM16 must be a literal: steady state leaves the next tile in
        // flight (cnt<=1); last iteration drains (cnt<=0).
        if (c + 1 < nk) __builtin_amdgcn_s_wait_tensorcnt((short)1);
        else            __builtin_amdgcn_s_wait_tensorcnt((short)0);
        __syncthreads();
        const _Float16* Ab = &As[c & 1][0];
#else
    for (int c = 0; c < nk; ++c) {
        // fallback: cooperative global->LDS staging (each thread copies 16B)
        const int crow = tid >> 2, cseg = (tid & 3) << 3;
        v8h tmp = *(const v8h*)(A + (size_t)(mblk + crow) * Kd + c * 32 + cseg);
        __syncthreads();
        *(v8h*)&As[0][crow * 32 + cseg] = tmp;
        __syncthreads();
        const _Float16* Ab = &As[0][0];
#endif
        const v16h a0 = lds_a_frag(Ab, m0l,      lane);
        const v16h a1 = lds_a_frag(Ab, m0l + 16, lane);
        const _Float16* bp = bp0 + ((size_t)c << 9);
        __builtin_prefetch(bp + (1u << 9), 0, 0);      // next k-chunk of B
        const v16h b0 = *(const v16h*)bp;
        const v16h b1 = *(const v16h*)(bp + bTileStride);
        acc[0][0] = wmma_f16(a0, b0, acc[0][0]);
        acc[0][1] = wmma_f16(a0, b1, acc[0][1]);
        acc[1][0] = wmma_f16(a1, b0, acc[1][0]);
        acc[1][1] = wmma_f16(a1, b1, acc[1][1]);
#if USE_TDM
        __syncthreads();                               // protect buffer reuse
#endif
    }

#pragma unroll
    for (int mi = 0; mi < 2; ++mi)
#pragma unroll
        for (int ni = 0; ni < 2; ++ni) {
            const int col = n0 + ni * 16 + (lane & 15);
#pragma unroll
            for (int r = 0; r < 8; ++r) {
                const int row = mblk + m0l + mi * 16 + r + ((lane >> 4) << 3);
                const float v = acc[mi][ni][r];
                if (EPI == 0) {
                    Yf[(size_t)row * N + col] = v;
                } else if (EPI == 1) {
                    Yf[(size_t)row * N + col] += v;
                } else {
                    const float g = 0.5f * v * (1.0f + erff(v * 0.70710678118654752f));
                    Yh[(size_t)row * N + col] = (_Float16)g;
                }
            }
        }
}

// ---------------------------------------------------------------------------
// Patch embedding: h[t][d] = sum_p x[t*32+p] * in_w[d][p] + in_b[d]
// ---------------------------------------------------------------------------
__global__ __launch_bounds__(256) void patch_embed_k(
        const float* __restrict__ x, const float* __restrict__ iw,
        const float* __restrict__ ib, float* __restrict__ h) {
    const int idx = blockIdx.x * 256 + threadIdx.x;   // t*512 + d
    const int t = idx >> 9, d = idx & 511;
    const float* xr = x + t * PATCH;
    const float* wr = iw + d * PATCH;
    float s = ib[d];
#pragma unroll
    for (int p = 0; p < PATCH; ++p) s += xr[p] * wr[p];
    h[idx] = s;
}

// ---------------------------------------------------------------------------
// LayerNorm over 512 dims: one block per token row, f16 output (GEMM A input)
// ---------------------------------------------------------------------------
__global__ __launch_bounds__(256) void layernorm_k(
        const float* __restrict__ h, const float* __restrict__ g,
        const float* __restrict__ b, _Float16* __restrict__ out) {
    __shared__ float red[256];
    const int row = blockIdx.x, tid = threadIdx.x;
    const float* xr = h + (size_t)row * DIM;
    const float v0 = xr[tid], v1 = xr[tid + 256];

    red[tid] = v0 + v1;
    for (int s = 128; s > 0; s >>= 1) {
        __syncthreads();
        if (tid < s) red[tid] += red[tid + s];
    }
    __syncthreads();
    const float mean = red[0] * (1.0f / DIM);
    __syncthreads();

    const float d0 = v0 - mean, d1 = v1 - mean;
    red[tid] = d0 * d0 + d1 * d1;
    for (int s = 128; s > 0; s >>= 1) {
        __syncthreads();
        if (tid < s) red[tid] += red[tid + s];
    }
    __syncthreads();
    const float rstd = rsqrtf(red[0] * (1.0f / DIM) + 1e-5f);

    out[(size_t)row * DIM + tid]       = (_Float16)(d0 * rstd * g[tid] + b[tid]);
    out[(size_t)row * DIM + tid + 256] = (_Float16)(d1 * rstd * g[tid + 256] + b[tid + 256]);
}

// ---------------------------------------------------------------------------
// Neighborhood attention (window K=127): one wave32 per (token, head).
// lane l owns dims {l, l+32}; logit via shfl_xor wave reduction; online softmax.
// ---------------------------------------------------------------------------
__global__ __launch_bounds__(256) void natten_k(
        const float* __restrict__ qkv, const float* __restrict__ rpb,
        _Float16* __restrict__ attno) {
    const int lane = threadIdx.x & 31;
    const int wave = threadIdx.x >> 5;
    const int pair = blockIdx.x * 8 + wave;     // [0, SEQ*HEADS)
    const int i = pair >> 3;                    // token
    const int hd = pair & 7;                    // head

    const float* qr = qkv + (size_t)i * (3 * DIM) + hd * DH;
    const float q0 = qr[lane] * 0.125f;         // DH^-0.5
    const float q1 = qr[lane + 32] * 0.125f;

    int ni = i - (KWIN / 2);
    ni = ni < 0 ? 0 : ni;
    ni = ni > (SEQ - KWIN) ? (SEQ - KWIN) : ni;
    const float* rp = rpb + (size_t)hd * (2 * KWIN - 1) + (KWIN - 1 - (i - ni));

    float m = -1e30f, s = 0.0f, o0 = 0.0f, o1 = 0.0f;
    for (int k = 0; k < KWIN; ++k) {
        const int j = ni + k;
        const float* kr = qkv + (size_t)j * (3 * DIM) + DIM + hd * DH;
        float part = q0 * kr[lane] + q1 * kr[lane + 32];
#pragma unroll
        for (int off = 16; off > 0; off >>= 1) part += __shfl_xor(part, off, 32);
        const float logit = part + rp[k];
        const float mn = fmaxf(m, logit);
        const float c = __expf(m - mn);
        const float p = __expf(logit - mn);
        s = s * c + p;
        const float* vr = qkv + (size_t)j * (3 * DIM) + 2 * DIM + hd * DH;
        o0 = o0 * c + p * vr[lane];
        o1 = o1 * c + p * vr[lane + 32];
        m = mn;
    }
    const float inv = 1.0f / s;
    attno[(size_t)i * DIM + hd * DH + lane]      = (_Float16)(o0 * inv);
    attno[(size_t)i * DIM + hd * DH + lane + 32] = (_Float16)(o1 * inv);
}

// ---------------------------------------------------------------------------
// Output projection: y[t*32+p] = tanh(h[t,:] . out_w[p,:] + out_b[p])
// ---------------------------------------------------------------------------
__global__ __launch_bounds__(256) void out_proj_k(
        const float* __restrict__ h, const float* __restrict__ ow,
        const float* __restrict__ ob, float* __restrict__ y) {
    const int lane = threadIdx.x & 31;
    const int wave = threadIdx.x >> 5;
    const int o = blockIdx.x * 8 + wave;        // [0, 16384)
    const int t = o >> 5, p = o & 31;
    const float* hr = h + (size_t)t * DIM;
    const float* wr = ow + (size_t)p * DIM;
    float part = 0.0f;
#pragma unroll 4
    for (int d = lane; d < DIM; d += 32) part += hr[d] * wr[d];
#pragma unroll
    for (int off = 16; off > 0; off >>= 1) part += __shfl_xor(part, off, 32);
    if (lane == 0) y[o] = tanhf(part + ob[p]);
}

// ---------------------------------------------------------------------------
// Host orchestration — all launches on `stream`, graph-capture safe.
// ---------------------------------------------------------------------------
extern "C" void kernel_launch(void* const* d_in, const int* in_sizes, int n_in,
                              void* d_out, int out_size, void* d_ws, size_t ws_size,
                              hipStream_t stream) {
    const float* x      = (const float*)d_in[0];
    const float* in_w   = (const float*)d_in[1];
    const float* in_b   = (const float*)d_in[2];
    const float* ln1_g  = (const float*)d_in[3];
    const float* ln1_b  = (const float*)d_in[4];
    const float* qkv_w  = (const float*)d_in[5];
    const float* qkv_b  = (const float*)d_in[6];
    const float* rpb    = (const float*)d_in[7];
    const float* proj_w = (const float*)d_in[8];
    const float* proj_b = (const float*)d_in[9];
    const float* ln2_g  = (const float*)d_in[10];
    const float* ln2_b  = (const float*)d_in[11];
    const float* ff1_w  = (const float*)d_in[12];
    const float* ff1_b  = (const float*)d_in[13];
    const float* ff2_w  = (const float*)d_in[14];
    const float* ff2_b  = (const float*)d_in[15];
    const float* out_w  = (const float*)d_in[16];
    const float* out_b  = (const float*)d_in[17];

    // ---- workspace layout (~55.3 MB) ----
    char* ws = (char*)d_ws;
    float*    h     = (float*)ws;                                        // 1 MB
    float*    qkv   = (float*)(ws + (1u << 20));                         // 3 MB
    _Float16* xn    = (_Float16*)(ws + 4u * (1u << 20));                 // 0.5 MB
    _Float16* attno = (_Float16*)(ws + 4u * (1u << 20) + SEQ * DIM * 2); // 0.5 MB
    _Float16* z     = (_Float16*)(ws + 5u * (1u << 20));                 // 2 MB
    _Float16* pw    = (_Float16*)(ws + 7u * (1u << 20));                 // 48 MB
    _Float16* pqkv  = pw;                                    // 8*1536*512
    _Float16* pproj = pqkv  + (size_t)LAYERS * 3 * DIM * DIM;
    _Float16* pff1  = pproj + (size_t)LAYERS * DIM * DIM;
    _Float16* pff2  = pff1  + (size_t)LAYERS * FFDIM * DIM;

    // ---- pre-pack weights to f16 fragment-major (L2-resident) ----
    for (int l = 0; l < LAYERS; ++l) {
        pack_w_k<<<(3 * DIM * DIM / 16) / 256, 256, 0, stream>>>(
            qkv_w + (size_t)l * 3 * DIM * DIM, pqkv + (size_t)l * 3 * DIM * DIM,
            3 * DIM, DIM);
        pack_w_k<<<(DIM * DIM / 16) / 256, 256, 0, stream>>>(
            proj_w + (size_t)l * DIM * DIM, pproj + (size_t)l * DIM * DIM,
            DIM, DIM);
        pack_w_k<<<(FFDIM * DIM / 16) / 256, 256, 0, stream>>>(
            ff1_w + (size_t)l * FFDIM * DIM, pff1 + (size_t)l * FFDIM * DIM,
            FFDIM, DIM);
        pack_w_k<<<(DIM * FFDIM / 16) / 256, 256, 0, stream>>>(
            ff2_w + (size_t)l * DIM * FFDIM, pff2 + (size_t)l * DIM * FFDIM,
            DIM, FFDIM);
    }

    patch_embed_k<<<(SEQ * DIM) / 256, 256, 0, stream>>>(x, in_w, in_b, h);

    for (int l = 0; l < LAYERS; ++l) {
        // --- attention block ---
        layernorm_k<<<SEQ, 256, 0, stream>>>(h, ln1_g + l * DIM, ln1_b + l * DIM, xn);
        gemm_wmma<0><<<dim3((3 * DIM) / 128, SEQ / 64), 256, 0, stream>>>(
            xn, pqkv + (size_t)l * 3 * DIM * DIM, qkv_b + (size_t)l * 3 * DIM,
            qkv, nullptr, SEQ, 3 * DIM, DIM);
        natten_k<<<(SEQ * HEADS) / 8, 256, 0, stream>>>(
            qkv, rpb + (size_t)l * HEADS * (2 * KWIN - 1), attno);
        gemm_wmma<1><<<dim3(DIM / 128, SEQ / 64), 256, 0, stream>>>(
            attno, pproj + (size_t)l * DIM * DIM, proj_b + (size_t)l * DIM,
            h, nullptr, SEQ, DIM, DIM);

        // --- MLP block ---
        layernorm_k<<<SEQ, 256, 0, stream>>>(h, ln2_g + l * DIM, ln2_b + l * DIM, xn);
        gemm_wmma<2><<<dim3(FFDIM / 128, SEQ / 64), 256, 0, stream>>>(
            xn, pff1 + (size_t)l * FFDIM * DIM, ff1_b + (size_t)l * FFDIM,
            nullptr, z, SEQ, FFDIM, DIM);
        gemm_wmma<1><<<dim3(DIM / 128, SEQ / 64), 256, 0, stream>>>(
            z, pff2 + (size_t)l * DIM * FFDIM, ff2_b + (size_t)l * DIM,
            h, nullptr, SEQ, DIM, FFDIM);
    }

    out_proj_k<<<(SEQ * PATCH) / 8, 256, 0, stream>>>(h, out_w, out_b, (float*)d_out);
}
